// BackBone_25091198943727
// MI455X (gfx1250) — matrix operand order repaired and checked by reading, not compiled, not run on hardware
//
#include <hip/hip_runtime.h>
#include <hip/hip_bf16.h>

// ---------------- problem constants ----------------
#define BB   64
#define TT   512
#define CC   32
#define NPP  8
#define RTT  128
#define RR   16          // ref points per patch
#define DD   128
#define HH   4
#define NLL  2
#define FLL  3
#define FHH  8
#define DFFN 256
#define PLN  96
#define LL   (TT + RR)   // 528
#define BL   (BB * LL)   // 33792
#define LKPAD 544        // 17 chunks of 32 (>= LL)
#define REF_STEP (1.0f / 127.0f)
#define PATCH_W  (1.0f / 8.0f)

typedef __attribute__((ext_vector_type(16))) __bf16 v16bf;
typedef __attribute__((ext_vector_type(8)))  float  v8f;

union ABu { v16bf v; unsigned int u[8]; };

static __device__ __forceinline__ unsigned short f2bf(float f) {
    unsigned int x = __float_as_uint(f);
    unsigned int r = (x + 0x7FFFu + ((x >> 16) & 1u)) >> 16;
    return (unsigned short)r;
}

// =====================================================================
// 0) weight pre-pack: f32 (K,N) row-major -> bf16-pair dwords laid out
//    in the same k order the GEMM loads A (k = 16*half + s):
//    Wsw[((nt*(K/32) + kt)*32 + lane)*8 + v] packs rows {kt*32+2v+16*half,
//    +1} at column nt*16+(lane&15).  Done once per launch; inner GEMM
//    loops become two contiguous b128 loads with zero conversion VALU.
// =====================================================================
__global__ void k_pack_w(const float* __restrict__ W,
                         unsigned int* __restrict__ Wsw, int K, int N)
{
    int idx = blockIdx.x * blockDim.x + threadIdx.x;
    int total = (N / 16) * (K / 32) * 256;
    if (idx >= total) return;
    int v    = idx & 7;
    int lane = (idx >> 3) & 31;
    int kt   = (idx >> 8) % (K / 32);
    int nt   = (idx >> 8) / (K / 32);
    int half = lane >> 4, rloc = lane & 15;
    int n  = nt * 16 + rloc;
    int kk = kt * 32 + 2 * v + half * 16;
    unsigned int lo = f2bf(W[(size_t)kk * N + n]);
    unsigned int hi = f2bf(W[(size_t)(kk + 1) * N + n]);
    Wsw[idx] = lo | (hi << 16);
}

// =====================================================================
// 1) split patch: exact stable compaction + ref insert + time sort
// =====================================================================
__global__ void k_split_patch(const float* __restrict__ vals,
                              const float* __restrict__ mask,
                              const float* __restrict__ time,
                              int p,
                              unsigned short* __restrict__ v_bf,
                              float* __restrict__ ptime,
                              int*   __restrict__ km,
                              int*   __restrict__ isrp)
{
    int b = blockIdx.x;
    __shared__ float s_time[TT];
    __shared__ int   s_tm[TT];
    float start = p * PATCH_W, end = (p + 1) * PATCH_W;

    for (int t = threadIdx.x; t < TT; t += blockDim.x) {
        float tv = time[b * TT + t];
        const float* mrow = mask + ((size_t)b * TT + t) * CC;
        float ms = 0.f;
        for (int c = 0; c < CC; ++c) ms += mrow[c];
        s_time[t] = tv;
        s_tm[t] = (tv >= start && tv <= end && ms > 0.f) ? 1 : 0;
    }
    __syncthreads();

    for (int l = threadIdx.x; l < LL; l += blockDim.x) {
        ptime[(size_t)b * LL + l] = 0.f;
        km[(size_t)b * LL + l] = 0;
        isrp[(size_t)b * LL + l] = 0;
    }
    for (int j = threadIdx.x; j < LL * CC; j += blockDim.x)
        v_bf[(size_t)b * LL * CC + j] = 0;
    __syncthreads();

    for (int t = threadIdx.x; t < TT; t += blockDim.x) {
        if (!s_tm[t]) continue;
        float ti = s_time[t];
        int rank = 0;
        for (int j = 0; j < TT; ++j)
            if (s_tm[j] && (s_time[j] < ti || (s_time[j] == ti && j < t))) rank++;
        for (int r = 0; r < RR; ++r) {
            float rt = (float)(p * RR + r) * REF_STEP;
            if (rt < ti) rank++;
        }
        int pos = rank;
        ptime[(size_t)b * LL + pos] = ti;
        km[(size_t)b * LL + pos] = 1;
        const float* vrow = vals + ((size_t)b * TT + t) * CC;
        const float* mrow = mask + ((size_t)b * TT + t) * CC;
        unsigned short* dst = v_bf + ((size_t)b * LL + pos) * CC;
        for (int c = 0; c < CC; ++c) dst[c] = f2bf(vrow[c] * mrow[c]);
    }
    for (int r = threadIdx.x; r < RR; r += blockDim.x) {
        float rt = (float)(p * RR + r) * REF_STEP;
        int cnt = 0;
        for (int j = 0; j < TT; ++j)
            if (s_tm[j] && s_time[j] <= rt) cnt++;
        int pos = cnt + r;
        ptime[(size_t)b * LL + pos] = rt;
        km[(size_t)b * LL + pos] = 1;
        isrp[(size_t)b * LL + pos] = 1;
    }
}

// =====================================================================
// 2) time embedding: te = bf16(sin(ptime*tw + tb)), (BL, D)
// =====================================================================
__global__ void k_time_embed(const float* __restrict__ ptime,
                             const float* __restrict__ tw,
                             const float* __restrict__ tb,
                             unsigned short* __restrict__ te)
{
    int idx = blockIdx.x * blockDim.x + threadIdx.x;
    if (idx >= BL * DD) return;
    int d  = idx & (DD - 1);
    int bl = idx >> 7;
    te[idx] = f2bf(sinf(ptime[bl] * tw[d] + tb[d]));
}

// =====================================================================
// 3) WMMA GEMM with pre-packed weights.
//    4 waves/block; each wave owns a 16-row strip and all NT=N/16 column
//    tiles: A loaded once per K-step, NT independent back-to-back WMMAs.
// =====================================================================
template <int NT, bool BIAS, bool RESID, bool OUTF, bool OUTB, int ACT>
__global__ void k_gemm_sw(const unsigned short* __restrict__ A,
                          const unsigned int* __restrict__ Wsw,
                          const float* __restrict__ bias,
                          const float* __restrict__ resid,
                          float* __restrict__ outF,
                          unsigned short* __restrict__ outB,
                          int M, int K, int N)
{
    int wave = threadIdx.x >> 5;
    int lane = threadIdx.x & 31;
    int mt = blockIdx.x * 4 + wave;
    int half = lane >> 4, rloc = lane & 15;
    v8f zero = {};
    v8f acc[NT];
#pragma unroll
    for (int j = 0; j < NT; ++j) acc[j] = zero;
    int kt32 = K >> 5;
    const unsigned short* Arow = A + (size_t)(mt * 16 + rloc) * K;
    for (int kt = 0; kt < kt32; ++kt) {
        ABu a;
        const unsigned int* arow = (const unsigned int*)(Arow + kt * 32);
        int kb = half * 16;
#pragma unroll
        for (int i = 0; i < 4; ++i) a.u[i]     = arow[(kb + 2 * i) >> 1];
#pragma unroll
        for (int i = 0; i < 4; ++i) a.u[4 + i] = arow[(8 + kb + 2 * i) >> 1];
        if (kt + 1 < kt32)
            __builtin_prefetch(Arow + (kt + 1) * 32, 0, 3);  // global_prefetch_b8
#pragma unroll
        for (int j = 0; j < NT; ++j) {
            ABu bw;
            const unsigned int* bp = Wsw + ((size_t)(j * kt32 + kt) * 32 + lane) * 8;
#pragma unroll
            for (int v = 0; v < 8; ++v) bw.u[v] = bp[v];
            acc[j] = __builtin_amdgcn_wmma_f32_16x16x32_bf16(
                false, a.v, false, bw.v, (short)0, acc[j], false, false);
        }
    }
#pragma unroll
    for (int j = 0; j < NT; ++j) {
        int n = j * 16 + rloc;
#pragma unroll
        for (int i = 0; i < 8; ++i) {
            int row = mt * 16 + i + half * 8;
            size_t o = (size_t)row * N + n;
            float x = acc[j][i];
            if (BIAS)  x += bias[n];
            if (RESID) x += resid[o];
            if (ACT == 1) x = fmaxf(x, 0.f);
            if (OUTF) outF[o] = x;
            if (OUTB) outB[o] = f2bf(x);
        }
    }
}

// =====================================================================
// 4) encoder attention, one wave per (q-tile, head, batch)
// =====================================================================
__global__ void k_attn(const unsigned short* __restrict__ qb,
                       const unsigned short* __restrict__ kbuf,
                       const unsigned short* __restrict__ vb,
                       const int* __restrict__ km,
                       unsigned short* __restrict__ ob)
{
    int qt = blockIdx.x, h = blockIdx.y, b = blockIdx.z;
    int lane = threadIdx.x;
    int half = lane >> 4;
    int rloc = lane & 15;
    __shared__ float s_sc[16 * LKPAD];
    __shared__ float s_mx2[32], s_sm2[32];

    for (int j = lane; j < 16 * (LKPAD - LL); j += 32) {
        int r = j / (LKPAD - LL), cc = j % (LKPAD - LL);
        s_sc[r * LKPAD + LL + cc] = -1e9f;
    }

    ABu aq;
    {
        const unsigned int* arow = (const unsigned int*)(
            qb + ((size_t)b * LL + qt * 16 + rloc) * DD + h * 32);
        int kb = half * 16;
#pragma unroll
        for (int i = 0; i < 4; ++i) aq.u[i]     = arow[(kb + 2 * i) >> 1];
#pragma unroll
        for (int i = 0; i < 4; ++i) aq.u[4 + i] = arow[(8 + kb + 2 * i) >> 1];
    }
    const float scale = 0.17677669529663687f;  // 1/sqrt(32)

    for (int kt = 0; kt < LL / 16; ++kt) {
        ABu bk;
        int kcol = kt * 16 + rloc;
        const unsigned int* kr32 = (const unsigned int*)(
            kbuf + ((size_t)b * LL + kcol) * DD + h * 32);
#pragma unroll
        for (int v = 0; v < 8; ++v) {
            int kk = 2 * v + half * 16;
            bk.u[v] = kr32[kk >> 1];
        }
        v8f s = {};
        s = __builtin_amdgcn_wmma_f32_16x16x32_bf16(false, aq.v, false, bk.v,
                                                    (short)0, s, false, false);
#pragma unroll
        for (int i = 0; i < 8; ++i) {
            int r = i + half * 8;
            float sc = s[i] * scale;
            if (!km[(size_t)b * LL + kcol]) sc = -1e9f;
            s_sc[r * LKPAD + kcol] = sc;
        }
    }
    __syncthreads();

    // row stats, 2 lanes per row (halves of the key axis)
    {
        float m = -1e30f;
        for (int j = half * (LKPAD / 2); j < (half + 1) * (LKPAD / 2); ++j)
            m = fmaxf(m, s_sc[rloc * LKPAD + j]);
        s_mx2[lane] = m;
    }
    __syncthreads();
    {
        float m = fmaxf(s_mx2[rloc], s_mx2[16 + rloc]);
        float su = 0.f;
        for (int j = half * (LKPAD / 2); j < (half + 1) * (LKPAD / 2); ++j)
            su += expf(s_sc[rloc * LKPAD + j] - m);
        s_sm2[lane] = su;
    }
    __syncthreads();
    float m   = fmaxf(s_mx2[rloc], s_mx2[16 + rloc]);
    float inv = 1.f / (s_sm2[rloc] + s_sm2[16 + rloc]);

    v8f c0 = {}, c1 = {};
    for (int kc = 0; kc < LKPAD / 32; ++kc) {
        ABu ap, b0, b1;
        int kb = half * 16;
#pragma unroll
        for (int i = 0; i < 8; ++i) {
            int klo = kb + 2 * i;
            int col = kc * 32 + klo;
            float p0 = expf(s_sc[rloc * LKPAD + col]     - m) * inv;
            float p1 = expf(s_sc[rloc * LKPAD + col + 1] - m) * inv;
            ap.u[i] = (unsigned int)f2bf(p0) | ((unsigned int)f2bf(p1) << 16);
        }
#pragma unroll
        for (int v = 0; v < 8; ++v) {
            int kk = kc * 32 + 2 * v + half * 16;
            unsigned short e0a = 0, e1a = 0, e0b = 0, e1b = 0;
            if (kk + 1 < LL) {
                size_t r0 = ((size_t)b * LL + kk) * DD + h * 32;
                size_t r1 = ((size_t)b * LL + kk + 1) * DD + h * 32;
                e0a = vb[r0 + rloc];      e1a = vb[r1 + rloc];
                e0b = vb[r0 + 16 + rloc]; e1b = vb[r1 + 16 + rloc];
            }
            b0.u[v] = (unsigned int)e0a | ((unsigned int)e1a << 16);
            b1.u[v] = (unsigned int)e0b | ((unsigned int)e1b << 16);
        }
        c0 = __builtin_amdgcn_wmma_f32_16x16x32_bf16(false, ap.v, false, b0.v,
                                                     (short)0, c0, false, false);
        c1 = __builtin_amdgcn_wmma_f32_16x16x32_bf16(false, ap.v, false, b1.v,
                                                     (short)0, c1, false, false);
    }
#pragma unroll
    for (int i = 0; i < 8; ++i) {
        int row = qt * 16 + i + half * 8;
        size_t base = ((size_t)b * LL + row) * DD + h * 32;
        ob[base + rloc]      = f2bf(c0[i]);
        ob[base + 16 + rloc] = f2bf(c1[i]);
    }
}

// =====================================================================
// 5) gather ref-point rows into reps (B, NP, R, C)
// =====================================================================
__global__ void k_gather_reps(const float* __restrict__ henc,
                              const int* __restrict__ isrp,
                              float* __restrict__ reps, int p)
{
    int b = blockIdx.x;
    __shared__ int pos[RR];
    if (threadIdx.x == 0) {
        int r = 0;
        for (int l = 0; l < LL && r < RR; ++l)
            if (isrp[(size_t)b * LL + l]) pos[r++] = l;
        for (; r < RR; ++r) pos[r] = 0;
    }
    __syncthreads();
    for (int j = threadIdx.x; j < RR * CC; j += blockDim.x) {
        int r = j / CC, c = j % CC;
        reps[(((size_t)b * NPP + p) * RR + r) * CC + c] =
            henc[((size_t)b * LL + pos[r]) * CC + c];
    }
}

// =====================================================================
// 6) former: tiny transformer, one wave per (b*C+c) sequence
// =====================================================================
__global__ void k_former(const float* __restrict__ reps,
    const float* fq,  const float* fqb, const float* fk,  const float* fkb,
    const float* fv,  const float* fvb, const float* fo,  const float* fob,
    const float* fc1, const float* fc1b, const float* fc2, const float* fc2b,
    const float* ln1g, const float* ln1b, const float* ln2g, const float* ln2b,
    const float* lnfg, const float* lnfb,
    unsigned short* __restrict__ fx)
{
    int n = blockIdx.x;
    int b = n / CC, c = n % CC;
    int t = threadIdx.x;
    __shared__ float X[NPP][RR], Q[NPP][RR], Kk[NPP][RR], V[NPP][RR], O[NPP][RR];
    __shared__ float Y1[NPP][DFFN];
    __shared__ float mu[NPP], vr[NPP];

    for (int j = t; j < NPP * RR; j += 32) {
        int np = j / RR, r = j % RR;
        float dv = expf(-(float)((r / 2) * 2) * (logf(10000.f) / (float)RR));
        float pe = ((r & 1) == 0) ? sinf((float)np * dv) : cosf((float)np * dv);
        X[np][r] = reps[(((size_t)b * NPP + np) * RR + r) * CC + c] + pe;
    }
    __syncthreads();

    for (int l = 0; l < FLL; ++l) {
        for (int j = t; j < NPP * RR; j += 32) {
            int s = j / RR, d = j % RR;
            float aq = fqb[l * RR + d], ak = fkb[l * RR + d], av = fvb[l * RR + d];
            for (int r = 0; r < RR; ++r) {
                float xv = X[s][r];
                aq += xv * fq[((size_t)l * RR + r) * RR + d];
                ak += xv * fk[((size_t)l * RR + r) * RR + d];
                av += xv * fv[((size_t)l * RR + r) * RR + d];
            }
            Q[s][d] = aq; Kk[s][d] = ak; V[s][d] = av;
        }
        __syncthreads();
        for (int j = t; j < NPP * FHH; j += 32) {
            int s = j / FHH, hh = j % FHH;
            int d0 = hh * 2;
            float sc[NPP], m = -1e30f;
            for (int kq = 0; kq < NPP; ++kq) {
                sc[kq] = (Q[s][d0] * Kk[kq][d0] + Q[s][d0 + 1] * Kk[kq][d0 + 1])
                         * 0.70710678118f;
                m = fmaxf(m, sc[kq]);
            }
            float su = 0.f;
            for (int kq = 0; kq < NPP; ++kq) { sc[kq] = expf(sc[kq] - m); su += sc[kq]; }
            float o0 = 0.f, o1 = 0.f;
            for (int kq = 0; kq < NPP; ++kq) {
                o0 += sc[kq] * V[kq][d0];
                o1 += sc[kq] * V[kq][d0 + 1];
            }
            O[s][d0] = o0 / su; O[s][d0 + 1] = o1 / su;
        }
        __syncthreads();
        for (int j = t; j < NPP * RR; j += 32) {
            int s = j / RR, d = j % RR;
            float a = fob[l * RR + d];
            for (int r = 0; r < RR; ++r) a += O[s][r] * fo[((size_t)l * RR + r) * RR + d];
            Q[s][d] = X[s][d] + a;
        }
        __syncthreads();
        if (t < NPP) {
            float m2 = 0.f;
            for (int d = 0; d < RR; ++d) m2 += Q[t][d];
            m2 /= RR;
            float v2 = 0.f;
            for (int d = 0; d < RR; ++d) { float dd = Q[t][d] - m2; v2 += dd * dd; }
            mu[t] = m2; vr[t] = v2 / RR;
        }
        __syncthreads();
        for (int j = t; j < NPP * RR; j += 32) {
            int s = j / RR, d = j % RR;
            X[s][d] = (Q[s][d] - mu[s]) * rsqrtf(vr[s] + 1e-5f)
                      * ln1g[l * RR + d] + ln1b[l * RR + d];
        }
        __syncthreads();
        for (int j = t; j < NPP * DFFN; j += 32) {
            int s = j / DFFN, f = j % DFFN;
            float a = fc1b[l * DFFN + f];
            for (int r = 0; r < RR; ++r) a += X[s][r] * fc1[((size_t)l * RR + r) * DFFN + f];
            float a3 = a * a * a;
            Y1[s][f] = 0.5f * a * (1.f + tanhf(0.7978845608f * (a + 0.044715f * a3)));
        }
        __syncthreads();
        for (int j = t; j < NPP * RR; j += 32) {
            int s = j / RR, d = j % RR;
            float a = fc2b[l * RR + d];
            for (int f = 0; f < DFFN; ++f) a += Y1[s][f] * fc2[((size_t)l * DFFN + f) * RR + d];
            Q[s][d] = X[s][d] + a;
        }
        __syncthreads();
        if (t < NPP) {
            float m2 = 0.f;
            for (int d = 0; d < RR; ++d) m2 += Q[t][d];
            m2 /= RR;
            float v2 = 0.f;
            for (int d = 0; d < RR; ++d) { float dd = Q[t][d] - m2; v2 += dd * dd; }
            mu[t] = m2; vr[t] = v2 / RR;
        }
        __syncthreads();
        for (int j = t; j < NPP * RR; j += 32) {
            int s = j / RR, d = j % RR;
            X[s][d] = (Q[s][d] - mu[s]) * rsqrtf(vr[s] + 1e-5f)
                      * ln2g[l * RR + d] + ln2b[l * RR + d];
        }
        __syncthreads();
    }
    if (t < NPP) {
        float m2 = 0.f;
        for (int d = 0; d < RR; ++d) m2 += X[t][d];
        m2 /= RR;
        float v2 = 0.f;
        for (int d = 0; d < RR; ++d) { float dd = X[t][d] - m2; v2 += dd * dd; }
        mu[t] = m2; vr[t] = v2 / RR;
    }
    __syncthreads();
    for (int j = t; j < NPP * RR; j += 32) {
        int s = j / RR, d = j % RR;
        float xv = (X[s][d] - mu[s]) * rsqrtf(vr[s] + 1e-5f) * lnfg[d] + lnfb[d];
        fx[(size_t)n * (RR * NPP) + d * NPP + s] = f2bf(xv);
    }
}

// =====================================================================
// 7) output transpose: (B*C, PL) -> (B, PL, C)
// =====================================================================
__global__ void k_out_transpose(const float* __restrict__ fin,
                                float* __restrict__ outp)
{
    int idx = blockIdx.x * blockDim.x + threadIdx.x;
    if (idx >= BB * PLN * CC) return;
    int c  = idx % CC;
    int tt = (idx / CC) % PLN;
    int b  = idx / (CC * PLN);
    outp[idx] = fin[((size_t)b * CC + c) * PLN + tt];
}

// =====================================================================
// host-side orchestration
// =====================================================================
extern "C" void kernel_launch(void* const* d_in, const int* in_sizes, int n_in,
                              void* d_out, int out_size, void* d_ws, size_t ws_size,
                              hipStream_t stream)
{
    const float* vals = (const float*)d_in[0];
    const float* mask = (const float*)d_in[1];
    const float* timep = (const float*)d_in[2];
    const float* tw   = (const float*)d_in[3];
    const float* tb   = (const float*)d_in[4];
    const float* Wv_in = (const float*)d_in[5];
    const float* eWq  = (const float*)d_in[6];
    const float* eWk  = (const float*)d_in[7];
    const float* eWv  = (const float*)d_in[8];
    const float* eWp  = (const float*)d_in[9];
    const float* eW1  = (const float*)d_in[10];
    const float* eW2  = (const float*)d_in[11];
    const float* Wout = (const float*)d_in[12];
    const float* fq   = (const float*)d_in[13];
    const float* fqb  = (const float*)d_in[14];
    const float* fk   = (const float*)d_in[15];
    const float* fkb  = (const float*)d_in[16];
    const float* fv   = (const float*)d_in[17];
    const float* fvb  = (const float*)d_in[18];
    const float* fo   = (const float*)d_in[19];
    const float* fob  = (const float*)d_in[20];
    const float* fc1  = (const float*)d_in[21];
    const float* fc1b = (const float*)d_in[22];
    const float* fc2  = (const float*)d_in[23];
    const float* fc2b = (const float*)d_in[24];
    const float* ln1g = (const float*)d_in[25];
    const float* ln1b = (const float*)d_in[26];
    const float* ln2g = (const float*)d_in[27];
    const float* ln2b = (const float*)d_in[28];
    const float* lnfg = (const float*)d_in[29];
    const float* lnfb = (const float*)d_in[30];
    const float* Wlin = (const float*)d_in[31];
    const float* blin = (const float*)d_in[32];

    char* ws = (char*)d_ws;
    size_t off = 0;
    auto alloc = [&](size_t bytes) -> void* {
        off = (off + 255) & ~(size_t)255;
        void* p = ws + off;
        off += bytes;
        return p;
    };

    float* ptime = (float*)alloc((size_t)BL * 4);
    int*   km    = (int*)  alloc((size_t)BL * 4);
    int*   isrp  = (int*)  alloc((size_t)BL * 4);
    unsigned short* v_bf  = (unsigned short*)alloc((size_t)BL * CC * 2);
    unsigned short* te_bf = (unsigned short*)alloc((size_t)BL * DD * 2);
    float* h_f            = (float*)         alloc((size_t)BL * DD * 4);
    unsigned short* h_bf  = (unsigned short*)alloc((size_t)BL * DD * 2);
    unsigned short* q_bf  = (unsigned short*)alloc((size_t)BL * DD * 2);
    unsigned short* k_bf  = (unsigned short*)alloc((size_t)BL * DD * 2);
    unsigned short* vv_bf = (unsigned short*)alloc((size_t)BL * DD * 2);
    unsigned short* o_bf  = (unsigned short*)alloc((size_t)BL * DD * 2);
    unsigned short* t1_bf = (unsigned short*)alloc((size_t)BL * DD * 2);
    float* henc  = (float*)alloc((size_t)BL * CC * 4);
    float* reps  = (float*)alloc((size_t)BB * NPP * RR * CC * 4);
    unsigned short* fx = (unsigned short*)alloc((size_t)BB * CC * RR * NPP * 2);
    float* fin   = (float*)alloc((size_t)BB * CC * PLN * 4);

    // ---- pre-pack all weight matrices into WMMA B layout (bf16 pairs) ----
    auto pack = [&](const float* W, int K, int N) -> unsigned int* {
        int total = (N / 16) * (K / 32) * 256;
        unsigned int* dst = (unsigned int*)alloc((size_t)total * 4);
        k_pack_w<<<(total + 255) / 256, 256, 0, stream>>>(W, dst, K, N);
        return dst;
    };
    unsigned int* sw_vin = pack(Wv_in, CC, DD);
    unsigned int *sw_q[NLL], *sw_k[NLL], *sw_v[NLL], *sw_p[NLL], *sw_1[NLL], *sw_2[NLL];
    for (int l = 0; l < NLL; ++l) {
        sw_q[l] = pack(eWq + (size_t)l * DD * DD, DD, DD);
        sw_k[l] = pack(eWk + (size_t)l * DD * DD, DD, DD);
        sw_v[l] = pack(eWv + (size_t)l * DD * DD, DD, DD);
        sw_p[l] = pack(eWp + (size_t)l * DD * DD, DD, DD);
        sw_1[l] = pack(eW1 + (size_t)l * DD * DD, DD, DD);
        sw_2[l] = pack(eW2 + (size_t)l * DD * DD, DD, DD);
    }
    unsigned int* sw_out = pack(Wout, DD, CC);
    unsigned int* sw_lin = pack(Wlin, DD, PLN);

    const int GB = BL / 64;   // GEMM blocks for M=BL (4 waves x 16 rows)

    for (int p = 0; p < NPP; ++p) {
        k_split_patch<<<BB, 256, 0, stream>>>(vals, mask, timep, p,
                                              v_bf, ptime, km, isrp);
        k_time_embed<<<(BL * DD + 255) / 256, 256, 0, stream>>>(ptime, tw, tb, te_bf);
        // h = v @ Wv_in (K=32)
        k_gemm_sw<8, false, false, true, true, 0><<<GB, 128, 0, stream>>>(
            v_bf, sw_vin, nullptr, nullptr, h_f, h_bf, BL, CC, DD);
        for (int l = 0; l < NLL; ++l) {
            k_gemm_sw<8, false, false, false, true, 0><<<GB, 128, 0, stream>>>(
                te_bf, sw_q[l], nullptr, nullptr, nullptr, q_bf, BL, DD, DD);
            k_gemm_sw<8, false, false, false, true, 0><<<GB, 128, 0, stream>>>(
                te_bf, sw_k[l], nullptr, nullptr, nullptr, k_bf, BL, DD, DD);
            k_gemm_sw<8, false, false, false, true, 0><<<GB, 128, 0, stream>>>(
                h_bf, sw_v[l], nullptr, nullptr, nullptr, vv_bf, BL, DD, DD);
            k_attn<<<dim3(LL / 16, HH, BB), 32, 0, stream>>>(q_bf, k_bf, vv_bf, km, o_bf);
            // h = h + o @ eWp
            k_gemm_sw<8, false, true, true, true, 0><<<GB, 128, 0, stream>>>(
                o_bf, sw_p[l], nullptr, h_f, h_f, h_bf, BL, DD, DD);
            // t1 = relu(h @ eW1)
            k_gemm_sw<8, false, false, false, true, 1><<<GB, 128, 0, stream>>>(
                h_bf, sw_1[l], nullptr, nullptr, nullptr, t1_bf, BL, DD, DD);
            // h = h + t1 @ eW2
            k_gemm_sw<8, false, true, true, true, 0><<<GB, 128, 0, stream>>>(
                t1_bf, sw_2[l], nullptr, h_f, h_f, h_bf, BL, DD, DD);
        }
        // henc = h @ Wout (N=32)
        k_gemm_sw<2, false, false, true, false, 0><<<GB, 128, 0, stream>>>(
            h_bf, sw_out, nullptr, nullptr, henc, nullptr, BL, DD, CC);
        k_gather_reps<<<BB, 32, 0, stream>>>(henc, isrp, reps, p);
    }

    k_former<<<BB * CC, 32, 0, stream>>>(reps,
        fq, fqb, fk, fkb, fv, fvb, fo, fob, fc1, fc1b, fc2, fc2b,
        ln1g, ln1b, ln2g, ln2b, lnfg, lnfb, fx);

    // out = fx @ Wlin + blin, then transpose to (B, PL, C)
    k_gemm_sw<6, true, false, true, false, 0><<<(BB * CC) / 64, 128, 0, stream>>>(
        fx, sw_lin, blin, nullptr, fin, nullptr, BB * CC, RR * NPP, PLN);
    k_out_transpose<<<(BB * PLN * CC + 255) / 256, 256, 0, stream>>>(
        fin, (float*)d_out);
}
